// NystromAttention_63952063038011
// MI455X (gfx1250) — matrix operand order repaired and checked by reading, not compiled, not run on hardware
//
#include <hip/hip_runtime.h>
#include <hip/hip_bf16.h>

// ---------------------------------------------------------------------------
// Types for CDNA5 WMMA
// ---------------------------------------------------------------------------
typedef __attribute__((ext_vector_type(16))) __bf16        v16bf;
typedef __attribute__((ext_vector_type(8)))  float         v8f;
typedef __attribute__((ext_vector_type(2)))  float         v2f;
typedef __attribute__((ext_vector_type(8)))  unsigned int  v8u;

#if __has_builtin(__builtin_amdgcn_global_load_async_to_lds_b32)
#define HAVE_ASYNC_LDS 1
#define AS1 __attribute__((address_space(1)))
#define AS3 __attribute__((address_space(3)))
__device__ inline void async_copy_b32(const void* g, void* l) {
    __builtin_amdgcn_global_load_async_to_lds_b32((AS1 int*)g, (AS3 int*)l, 0, 0);
}
__device__ inline void wait_async0() {
#if __has_builtin(__builtin_amdgcn_s_wait_asynccnt)
    __builtin_amdgcn_s_wait_asynccnt(0);
#else
    asm volatile("s_wait_asynccnt 0x0" ::: "memory");
#endif
}
#else
#define HAVE_ASYNC_LDS 0
#endif

__device__ inline v8f v8f_zero() {
    v8f z;
    for (int i = 0; i < 8; ++i) z[i] = 0.0f;
    return z;
}

// bf16 <-> f32 helpers (bf16 stored as raw unsigned short)
__device__ inline unsigned short f2bf(float x) {
    unsigned int u = __float_as_uint(x);
    unsigned int r = (u + 0x7FFFu + ((u >> 16) & 1u)) >> 16;  // RNE
    return (unsigned short)r;
}
__device__ inline float bf2f(unsigned short b) {
    return __uint_as_float(((unsigned int)b) << 16);
}

__device__ inline float loadElem(const float* p)          { return *p; }
__device__ inline float loadElem(const unsigned short* p) { return bf2f(*p); }

// ---------------------------------------------------------------------------
// Problem constants
// ---------------------------------------------------------------------------
#define BB    4
#define HH    8
#define NN    4096
#define DH    64
#define DIM   512
#define MM    256
#define LL    16
#define BH    (BB*HH)           // 32
#define NDH   (NN*DH)           // 262144  per (b,h) q/k/v plane
#define MDH   (MM*DH)           // 16384   per (b,h) landmark plane
#define MMSQ  (MM*MM)           // 65536   per (b,h) 256x256 matrix

// ---------------------------------------------------------------------------
// Generic tiled bf16 GEMM (C = A[M,K] * B[K,N]), wave32 WMMA 16x16x32 bf16.
//   block: 256 threads = 8 waves, tile 64(M) x 128(N), K step 32.
//   Waves arranged 2(M) x 4(N), each wave owns a 32x32 subtile (2x2 WMMAs).
// A staged in LDS row-major [row][k], B staged transposed [col][k], so
// fragment reads are aligned dword loads matching ISA 7.12.2 layouts:
//   A lane (r=lane&15, half=lane>>4): vgpr g -> K = (g<4 ? half*8+2g : 16+half*8+2(g-4))
//   B lane (c=lane&15, half):         vgpr g -> K = (g<4 ? half*8+2g : 16+half*8+2(g-4))
//   D element: row = g + 8*half, col = lane&15
// When the source is already bf16 (and, for B, K-contiguous i.e. TRANSB),
// staging uses GLOBAL_LOAD_ASYNC_TO_LDS_B32 (ASYNCcnt) instead of a VGPR
// round trip: 2 bf16 elements per async op, per-lane global+LDS addresses.
// ---------------------------------------------------------------------------
#define TILE_M 64
#define TILE_N 128
#define TILE_K 32
#define LDS_LD 34   // padded K stride (elements)

template <typename TA, typename TB, bool TRANSB, typename Ep>
__global__ __launch_bounds__(256) void gemm_bf16_kernel(
    const TA* __restrict__ Ab, long long strideA, int lda,
    const TB* __restrict__ Bb, long long strideB, int ldb,
    int M, int N, int K, Ep ep)
{
    __shared__ unsigned short sA[TILE_M * LDS_LD];
    __shared__ unsigned short sB[TILE_N * LDS_LD];

    const int tid   = threadIdx.x;
    const int z     = blockIdx.z;
    const int mBase = blockIdx.x * TILE_M;
    const int nBase = blockIdx.y * TILE_N;

    const TA* A = Ab + (long long)z * strideA;
    const TB* B = Bb + (long long)z * strideB;

    const int lane  = tid & 31;
    const int wid   = tid >> 5;
    const int waveM = (wid >> 2) * 32;  // 0,32
    const int waveN = (wid & 3) * 32;   // 0..96
    const int r     = lane & 15;
    const int half  = lane >> 4;

    v8f acc[2][2];
    for (int i = 0; i < 2; ++i)
        for (int j = 0; j < 2; ++j) acc[i][j] = v8f_zero();

    for (int k0 = 0; k0 < K; k0 += TILE_K) {
        bool usedAsync = false;
        // ---- Stage A tile (64 x 32) ----
#if HAVE_ASYNC_LDS
        if constexpr (sizeof(TA) == 2) {
            // pure bf16 copy: 2 elements (4 bytes) per async op
            for (int e = tid; e < TILE_M * (TILE_K / 2); e += 256) {
                int row = e >> 4, ee = e & 15;
                int gr = mBase + row;
                unsigned short* lp = &sA[row * LDS_LD + 2 * ee];
                if (gr < M) {
                    async_copy_b32(A + (long long)gr * lda + (k0 + 2 * ee), lp);
                } else {
                    lp[0] = 0; lp[1] = 0;
                }
            }
            usedAsync = true;
        } else
#endif
        {
            if (k0 + TILE_K < K)
                __builtin_prefetch(A + (long long)mBase * lda + k0 + TILE_K, 0, 0);
            for (int e = tid; e < TILE_M * TILE_K; e += 256) {
                int row = e >> 5, kk = e & 31;
                int gr = mBase + row;
                float v = (gr < M) ? loadElem(A + (long long)gr * lda + (k0 + kk)) : 0.0f;
                sA[row * LDS_LD + kk] = f2bf(v);
            }
        }
        // ---- Stage B tile (32 x 128), stored transposed [col][k] ----
#if HAVE_ASYNC_LDS
        if constexpr (sizeof(TB) == 2 && TRANSB) {
            // K-contiguous in both global and LDS: 2 elements per async op
            for (int e = tid; e < TILE_N * (TILE_K / 2); e += 256) {
                int col = e >> 4, ee = e & 15;
                int gc = nBase + col;
                unsigned short* lp = &sB[col * LDS_LD + 2 * ee];
                if (gc < N) {
                    async_copy_b32(B + (long long)gc * ldb + (k0 + 2 * ee), lp);
                } else {
                    lp[0] = 0; lp[1] = 0;
                }
            }
            usedAsync = true;
        } else
#endif
        {
            for (int e = tid; e < TILE_K * TILE_N; e += 256) {
                int col = e & 127, kk = e >> 7;
                int gc = nBase + col;
                float v = 0.0f;
                if (gc < N) {
                    long long offb = TRANSB ? ((long long)gc * ldb + (k0 + kk))
                                            : ((long long)(k0 + kk) * ldb + gc);
                    v = loadElem(B + offb);
                }
                sB[col * LDS_LD + kk] = f2bf(v);
            }
        }
#if HAVE_ASYNC_LDS
        if (usedAsync) wait_async0();
#else
        (void)usedAsync;
#endif
        __syncthreads();

        v8u au[2], bu[2];
        for (int i = 0; i < 2; ++i) {
            const unsigned short* ab = &sA[(waveM + i * 16 + r) * LDS_LD];
            for (int g = 0; g < 8; ++g) {
                int kk = (g < 4) ? (half * 8 + 2 * g) : (16 + half * 8 + 2 * (g - 4));
                au[i][g] = *(const unsigned int*)(ab + kk);
            }
        }
        for (int j = 0; j < 2; ++j) {
            const unsigned short* bb = &sB[(waveN + j * 16 + r) * LDS_LD];
            for (int g = 0; g < 8; ++g) {
                int kk = (g < 4) ? (half * 8 + 2 * g) : (16 + half * 8 + 2 * (g - 4));
                bu[j][g] = *(const unsigned int*)(bb + kk);
            }
        }
        for (int i = 0; i < 2; ++i) {
            v16bf a = __builtin_bit_cast(v16bf, au[i]);
            for (int j = 0; j < 2; ++j) {
                v16bf b = __builtin_bit_cast(v16bf, bu[j]);
                acc[i][j] = __builtin_amdgcn_wmma_f32_16x16x32_bf16(
                    false, a, false, b, (short)0, acc[i][j], false, false);
            }
        }
        __syncthreads();
    }

    for (int i = 0; i < 2; ++i)
        for (int j = 0; j < 2; ++j)
            for (int g = 0; g < 8; ++g) {
                int row = mBase + waveM + i * 16 + half * 8 + g;
                int col = nBase + waveN + j * 16 + r;
                if (row < M && col < N) ep(z, row, col, acc[i][j][g]);
            }
}

// ---------------------------------------------------------------------------
// Epilogue functors
// ---------------------------------------------------------------------------
struct EpF32 {
    float* C; long long strideC; int ldc;
    __device__ void operator()(int z, int r, int c, float v) const {
        C[(long long)z * strideC + (long long)r * ldc + c] = v;
    }
};
struct EpBF16 {
    unsigned short* C; long long strideC; int ldc;
    __device__ void operator()(int z, int r, int c, float v) const {
        C[(long long)z * strideC + (long long)r * ldc + c] = f2bf(v);
    }
};
// QKV split: row in [0,16384) = b*4096+n ; col in [0,1536) = sec*512 + h*64 + d
// writes bf16 into [b][h][n][d] planes, scaling q by dh^-0.5.
struct EpQKV {
    unsigned short *qp, *kp, *vp; float scale;
    __device__ void operator()(int z, int r, int c, float v) const {
        (void)z;
        int sec = c >> 9, hd = c & 511;
        int h = hd >> 6, d = hd & 63;
        int b = r >> 12, n = r & 4095;
        long long idx = (((long long)(b * HH + h) * NN + n) * DH) + d;
        if (sec == 0)      qp[idx] = f2bf(v * scale);
        else if (sec == 1) kp[idx] = f2bf(v);
        else               vp[idx] = f2bf(v);
    }
};
// out2 (batched per z=b*8+h, [4096,64]) + conv residual -> hcat[b][n][h*64+d] bf16
struct EpOut2 {
    const float* res; unsigned short* hcat;
    __device__ void operator()(int z, int r, int c, float v) const {
        int b = z >> 3, h = z & 7;
        float val = v + res[(long long)z * NDH + (long long)r * DH + c];
        hcat[((long long)(b * NN + r) * DIM) + h * DH + c] = f2bf(val);
    }
};
// final GEMM: + bias, write f32 to d_out  (batch==1, [16384,512])
struct EpBias {
    const float* bias; float* out;
    __device__ void operator()(int z, int r, int c, float v) const {
        (void)z;
        out[(long long)r * DIM + c] = v + bias[c];
    }
};

// ---------------------------------------------------------------------------
// Batched fp32 WMMA matmul for pinv: D = op(A[256,256] @ B[256,256]) per z
//   mode 0: D = A@B    mode 1: D = alpha*I - A@B    mode 2: D = 0.25*A@B
// Uses V_WMMA_F32_16X16X4_F32. block 256 = 8 waves; tile 64x64;
// wave = 32(M) x 16(N) subtile (2 WMMAs per 4-K step).
// Tiles staged via GLOBAL_LOAD_ASYNC_TO_LDS_B32 when available (pure f32
// copies; per-lane addresses implement the B transpose during the copy).
// ---------------------------------------------------------------------------
__global__ __launch_bounds__(256) void pinv_mm_kernel(
    const float* __restrict__ Ab, const float* __restrict__ Bb,
    float* __restrict__ Db, float alpha, int mode)
{
    __shared__ float sA[64 * 33];
    __shared__ float sB[64 * 33];   // transposed: [col][k]

    const int tid = threadIdx.x, z = blockIdx.z;
    const float* A = Ab + (long long)z * MMSQ;
    const float* B = Bb + (long long)z * MMSQ;
    float*       D = Db + (long long)z * MMSQ;

    const int mBase = blockIdx.x * 64, nBase = blockIdx.y * 64;
    const int lane = tid & 31, wid = tid >> 5;
    const int waveM = (wid >> 2) * 32;  // 0,32
    const int waveN = (wid & 3) * 16;   // 0..48
    const int r = lane & 15, half = lane >> 4;

    v8f acc[2];
    acc[0] = v8f_zero(); acc[1] = v8f_zero();

    for (int k0 = 0; k0 < MM; k0 += 32) {
#if HAVE_ASYNC_LDS
        for (int e = tid; e < 64 * 32; e += 256) {
            int row = e >> 5, kk = e & 31;
            async_copy_b32(A + (long long)(mBase + row) * MM + (k0 + kk),
                           &sA[row * 33 + kk]);
        }
        for (int e = tid; e < 32 * 64; e += 256) {
            int col = e & 63, kk = e >> 6;
            async_copy_b32(B + (long long)(k0 + kk) * MM + (nBase + col),
                           &sB[col * 33 + kk]);
        }
        wait_async0();
#else
        for (int e = tid; e < 64 * 32; e += 256) {
            int row = e >> 5, kk = e & 31;
            sA[row * 33 + kk] = A[(long long)(mBase + row) * MM + (k0 + kk)];
        }
        for (int e = tid; e < 32 * 64; e += 256) {
            int col = e & 63, kk = e >> 6;
            sB[col * 33 + kk] = B[(long long)(k0 + kk) * MM + (nBase + col)];
        }
#endif
        __syncthreads();

        for (int k4 = 0; k4 < 32; k4 += 4) {
            v2f bf;
            bf.x = sB[(waveN + r) * 33 + k4 + 2 * half];
            bf.y = sB[(waveN + r) * 33 + k4 + 2 * half + 1];
            for (int i = 0; i < 2; ++i) {
                v2f af;
                int rowl = waveM + i * 16 + r;
                af.x = sA[rowl * 33 + k4 + 2 * half];
                af.y = sA[rowl * 33 + k4 + 2 * half + 1];
                acc[i] = __builtin_amdgcn_wmma_f32_16x16x4_f32(
                    false, af, false, bf, (short)0, acc[i], false, false);
            }
        }
        __syncthreads();
    }

    for (int i = 0; i < 2; ++i)
        for (int g = 0; g < 8; ++g) {
            int row = mBase + waveM + i * 16 + half * 8 + g;
            int col = nBase + waveN + r;
            float v = acc[i][g];
            if (mode == 1)      v = alpha * (row == col ? 1.0f : 0.0f) - v;
            else if (mode == 2) v = 0.25f * v;
            D[(long long)row * MM + col] = v;
        }
}

// per-(b,h) max |row-sum| and max |col-sum| of attn2 -> partial[0..31], [32..63]
__global__ __launch_bounds__(256) void pinv_norm_kernel(const float* __restrict__ a2,
                                                        float* __restrict__ partial)
{
    int z = blockIdx.x, t = threadIdx.x;
    const float* a = a2 + (long long)z * MMSQ;
    float rs = 0.0f, cs = 0.0f;
    for (int j = 0; j < MM; ++j) {
        rs += fabsf(a[(long long)t * MM + j]);
        cs += fabsf(a[(long long)j * MM + t]);
    }
    __shared__ float rr[256], cc[256];
    rr[t] = rs; cc[t] = cs; __syncthreads();
    for (int s = 128; s > 0; s >>= 1) {
        if (t < s) { rr[t] = fmaxf(rr[t], rr[t + s]); cc[t] = fmaxf(cc[t], cc[t + s]); }
        __syncthreads();
    }
    if (t == 0) { partial[z] = rr[0]; partial[32 + z] = cc[0]; }
}

// z0 = attn2^T / (globalRowMax * globalColMax)   (reference uses GLOBAL maxima)
__global__ __launch_bounds__(256) void pinv_init_z_kernel(const float* __restrict__ a2,
                                                          const float* __restrict__ partial,
                                                          float* __restrict__ z0)
{
    int zb = blockIdx.z;
    int e  = blockIdx.x * 256 + threadIdx.x;      // 0..65535
    float rmax = 0.0f, cmax = 0.0f;
    for (int i = 0; i < 32; ++i) {
        rmax = fmaxf(rmax, partial[i]);
        cmax = fmaxf(cmax, partial[32 + i]);
    }
    float inv = 1.0f / (rmax * cmax);
    int i = e >> 8, j = e & 255;
    z0[(long long)zb * MMSQ + e] = a2[(long long)zb * MMSQ + (long long)j * MM + i] * inv;
}

// D = alpha*I - S (elementwise, batched 256x256)
__global__ __launch_bounds__(256) void ew_alphaI_kernel(const float* __restrict__ S,
                                                        float* __restrict__ D, float alpha)
{
    int zb = blockIdx.z;
    int e  = blockIdx.x * 256 + threadIdx.x;
    int i = e >> 8, j = e & 255;
    long long idx = (long long)zb * MMSQ + e;
    D[idx] = alpha * (i == j ? 1.0f : 0.0f) - S[idx];
}

__global__ __launch_bounds__(256) void f32_to_bf16_kernel(const float* __restrict__ in,
                                                          unsigned short* __restrict__ out,
                                                          long long nElem)
{
    long long i = (long long)blockIdx.x * 256 + threadIdx.x;
    if (i < nElem) out[i] = f2bf(in[i]);
}

// ---------------------------------------------------------------------------
// Row softmax: in f32 [R, C] -> out (f32 or bf16). One block per row.
// ---------------------------------------------------------------------------
__device__ inline void storeVal(float* p, float v)          { *p = v; }
__device__ inline void storeVal(unsigned short* p, float v) { *p = f2bf(v); }

template <typename OUT>
__global__ __launch_bounds__(256) void softmax_rows_kernel(const float* __restrict__ in,
                                                           OUT* __restrict__ out, int C)
{
    long long row = blockIdx.x;
    const float* x = in + row * (long long)C;
    OUT*         y = out + row * (long long)C;
    __shared__ float red[256];
    int t = threadIdx.x;

    float mx = -3.0e38f;
    for (int j = t; j < C; j += 256) mx = fmaxf(mx, x[j]);
    red[t] = mx; __syncthreads();
    for (int s = 128; s > 0; s >>= 1) { if (t < s) red[t] = fmaxf(red[t], red[t + s]); __syncthreads(); }
    float m = red[0]; __syncthreads();

    float s = 0.0f;
    for (int j = t; j < C; j += 256) s += __expf(x[j] - m);
    red[t] = s; __syncthreads();
    for (int ss = 128; ss > 0; ss >>= 1) { if (t < ss) red[t] += red[t + ss]; __syncthreads(); }
    float inv = 1.0f / red[0];

    for (int j = t; j < C; j += 256) storeVal(y + j, __expf(x[j] - m) * inv);
}

// ---------------------------------------------------------------------------
// Landmarks: mean over groups of 16 tokens of q/k (bf16 in, bf16 out)
// ---------------------------------------------------------------------------
__global__ __launch_bounds__(256) void landmark_kernel(const unsigned short* __restrict__ qb,
                                                       const unsigned short* __restrict__ kb,
                                                       unsigned short* __restrict__ qlb,
                                                       unsigned short* __restrict__ klb)
{
    long long i = (long long)blockIdx.x * 256 + threadIdx.x;   // BH*MM*DH = 524288
    if (i >= (long long)BH * MM * DH) return;
    int d = (int)(i & 63);
    long long t = i >> 6;
    int mIdx = (int)(t & 255);
    long long bh = t >> 8;
    long long base = bh * NDH + (long long)mIdx * LL * DH + d;
    float sq = 0.0f, sk = 0.0f;
    for (int j = 0; j < LL; ++j) {
        sq += bf2f(qb[base + (long long)j * DH]);
        sk += bf2f(kb[base + (long long)j * DH]);
    }
    qlb[i] = f2bf(sq * (1.0f / LL));
    klb[i] = f2bf(sk * (1.0f / LL));
}

// ---------------------------------------------------------------------------
// Depthwise residual conv over sequence: res[bh,n,d] = sum_j v[bh,n+j-16,d]*kern[h,j]
// ---------------------------------------------------------------------------
__global__ __launch_bounds__(256) void conv_res_kernel(const unsigned short* __restrict__ vb,
                                                       const float* __restrict__ kern,
                                                       float* __restrict__ res)
{
    long long i = (long long)blockIdx.x * 256 + threadIdx.x;   // BH*NN*DH = 8388608
    if (i >= (long long)BH * NN * DH) return;
    int d = (int)(i & 63);
    long long t = i >> 6;
    int n = (int)(t & 4095);
    long long bh = t >> 12;
    int h = (int)(bh & 7);
    const unsigned short* vplane = vb + bh * NDH;
    float acc = 0.0f;
    for (int j = 0; j < 33; ++j) {
        int p = n + j - 16;
        if (p >= 0 && p < NN)
            acc += bf2f(vplane[(long long)p * DH + d]) * kern[h * 33 + j];
    }
    res[i] = acc;
}

// ---------------------------------------------------------------------------
// Host orchestration
// ---------------------------------------------------------------------------
extern "C" void kernel_launch(void* const* d_in, const int* in_sizes, int n_in,
                              void* d_out, int out_size, void* d_ws, size_t ws_size,
                              hipStream_t stream) {
    (void)in_sizes; (void)n_in; (void)out_size; (void)ws_size;
    const float* x     = (const float*)d_in[0];   // [4,4096,512]
    const float* w_qkv = (const float*)d_in[1];   // [512,1536]
    const float* w_out = (const float*)d_in[2];   // [512,512]
    const float* b_out = (const float*)d_in[3];   // [512]
    const float* rker  = (const float*)d_in[4];   // [8,1,33,1]
    float* out = (float*)d_out;                   // [4,4096,512]

    char* base = (char*)d_ws;
    size_t off = 0;
    auto alloc = [&](size_t bytes) -> void* {
        void* p = base + off;
        off += (bytes + 255) & ~(size_t)255;
        return p;
    };

    unsigned short* qb     = (unsigned short*)alloc((size_t)BH * NDH * 2);
    unsigned short* kb     = (unsigned short*)alloc((size_t)BH * NDH * 2);
    unsigned short* vb     = (unsigned short*)alloc((size_t)BH * NDH * 2);
    unsigned short* qlb    = (unsigned short*)alloc((size_t)BH * MDH * 2);
    unsigned short* klb    = (unsigned short*)alloc((size_t)BH * MDH * 2);
    float*          logitsF= (float*)alloc((size_t)BH * NN * MM * 4);   // reused attn1 / attn3
    unsigned short* attn1b = (unsigned short*)alloc((size_t)BH * NN * MM * 2);
    unsigned short* attn3b = (unsigned short*)alloc((size_t)BH * MM * NN * 2);
    float*          attn2  = (float*)alloc((size_t)BH * MMSQ * 4);
    float*          z0     = (float*)alloc((size_t)BH * MMSQ * 4);
    float*          z1     = (float*)alloc((size_t)BH * MMSQ * 4);
    float*          xz     = (float*)alloc((size_t)BH * MMSQ * 4);
    float*          tb     = (float*)alloc((size_t)BH * MMSQ * 4);
    float*          ub     = (float*)alloc((size_t)BH * MMSQ * 4);
    unsigned short* zbb    = (unsigned short*)alloc((size_t)BH * MMSQ * 2);
    unsigned short* a3vb   = (unsigned short*)alloc((size_t)BH * MDH * 2);
    unsigned short* out1b  = (unsigned short*)alloc((size_t)BH * NN * MM * 2);
    float*          res    = (float*)alloc((size_t)BH * NDH * 4);
    unsigned short* hcat   = (unsigned short*)alloc((size_t)BB * NN * DIM * 2);
    float*          partial= (float*)alloc(64 * 4);

    // 1. fused QKV projection -> q/k/v bf16 in [b,h,n,dh] (q pre-scaled)
    {
        EpQKV ep{qb, kb, vb, 0.125f};   // dh^-0.5
        dim3 g((BB * NN) / TILE_M, (3 * DIM) / TILE_N, 1);
        gemm_bf16_kernel<float, float, false, EpQKV><<<g, 256, 0, stream>>>(
            x, 0, DIM, w_qkv, 0, 3 * DIM, BB * NN, 3 * DIM, DIM, ep);
    }
    // 2. landmarks
    landmark_kernel<<<(BH * MM * DH) / 256, 256, 0, stream>>>(qb, kb, qlb, klb);
    // 3. attn1 logits = q @ k_l^T   [z][4096][256]
    {
        EpF32 ep{logitsF, (long long)NN * MM, MM};
        dim3 g(NN / TILE_M, (MM + TILE_N - 1) / TILE_N, BH);
        gemm_bf16_kernel<unsigned short, unsigned short, true, EpF32><<<g, 256, 0, stream>>>(
            qb, NDH, DH, klb, MDH, DH, NN, MM, DH, ep);
    }
    // 4. softmax -> attn1 bf16
    softmax_rows_kernel<unsigned short><<<BH * NN, 256, 0, stream>>>(logitsF, attn1b, MM);
    // 5. attn2 logits = q_l @ k_l^T  (into xz scratch)
    {
        EpF32 ep{xz, MMSQ, MM};
        dim3 g(MM / TILE_M, (MM + TILE_N - 1) / TILE_N, BH);
        gemm_bf16_kernel<unsigned short, unsigned short, true, EpF32><<<g, 256, 0, stream>>>(
            qlb, MDH, DH, klb, MDH, DH, MM, MM, DH, ep);
    }
    // 6. softmax -> attn2 f32
    softmax_rows_kernel<float><<<BH * MM, 256, 0, stream>>>(xz, attn2, MM);
    // 7. attn3 logits = q_l @ k^T   [z][256][4096]
    {
        EpF32 ep{logitsF, (long long)MM * NN, NN};
        dim3 g(MM / TILE_M, NN / TILE_N, BH);
        gemm_bf16_kernel<unsigned short, unsigned short, true, EpF32><<<g, 256, 0, stream>>>(
            qlb, MDH, DH, kb, NDH, DH, MM, NN, DH, ep);
    }
    // 8. softmax -> attn3 bf16
    softmax_rows_kernel<unsigned short><<<BH * MM, 256, 0, stream>>>(logitsF, attn3b, NN);

    // 9. Moore-Penrose pinv of attn2 (fp32 WMMA 16x16x4)
    pinv_norm_kernel<<<BH, 256, 0, stream>>>(attn2, partial);
    pinv_init_z_kernel<<<dim3(MMSQ / 256, 1, BH), 256, 0, stream>>>(attn2, partial, z0);
    {
        auto mm = [&](const float* A, const float* B, float* D, float alpha, int mode) {
            pinv_mm_kernel<<<dim3(MM / 64, MM / 64, BH), 256, 0, stream>>>(A, B, D, alpha, mode);
        };
        float* zc = z0;
        float* zn = z1;
        for (int it = 0; it < 6; ++it) {
            mm(attn2, zc, xz, 0.0f, 0);                                        // xz = x @ z
            ew_alphaI_kernel<<<dim3(MMSQ / 256, 1, BH), 256, 0, stream>>>(xz, tb, 7.0f);   // t = 7I - xz
            mm(xz, tb, ub, 15.0f, 1);                                          // u = 15I - xz@t
            mm(xz, ub, tb, 13.0f, 1);                                          // t = 13I - xz@u
            mm(zc, tb, zn, 0.0f, 2);                                           // zn = 0.25 * z@t
            float* sw = zc; zc = zn; zn = sw;
        }
        // zc == z0 after 6 swaps
        f32_to_bf16_kernel<<<(BH * MMSQ) / 256, 256, 0, stream>>>(zc, zbb, (long long)BH * MMSQ);
    }

    // 11. attn3v = attn3 @ v   [z][256][64] bf16
    {
        EpBF16 ep{a3vb, MDH, DH};
        dim3 g(MM / TILE_M, 1, BH);
        gemm_bf16_kernel<unsigned short, unsigned short, false, EpBF16><<<g, 256, 0, stream>>>(
            attn3b, (long long)MM * NN, NN, vb, NDH, DH, MM, DH, NN, ep);
    }
    // 12. out1 = attn1 @ pinv   [z][4096][256] bf16
    {
        EpBF16 ep{out1b, (long long)NN * MM, MM};
        dim3 g(NN / TILE_M, (MM + TILE_N - 1) / TILE_N, BH);
        gemm_bf16_kernel<unsigned short, unsigned short, false, EpBF16><<<g, 256, 0, stream>>>(
            attn1b, (long long)NN * MM, MM, zbb, MMSQ, MM, NN, MM, MM, ep);
    }
    // 13. depthwise residual conv on v
    conv_res_kernel<<<(BH * NDH) / 256, 256, 0, stream>>>(vb, rker, res);
    // 14. out2 = out1 @ attn3v + res  -> hcat [b][n][h*64+d] bf16
    {
        EpOut2 ep{res, hcat};
        dim3 g(NN / TILE_M, 1, BH);
        gemm_bf16_kernel<unsigned short, unsigned short, false, EpOut2><<<g, 256, 0, stream>>>(
            out1b, (long long)NN * MM, MM, a3vb, MDH, DH, NN, DH, MM, ep);
    }
    // 15. final projection: out = hcat @ w_out + b_out (f32 out)
    {
        EpBias ep{b_out, out};
        dim3 g((BB * NN) / TILE_M, DIM / TILE_N, 1);
        gemm_bf16_kernel<unsigned short, float, false, EpBias><<<g, 256, 0, stream>>>(
            hcat, 0, DIM, w_out, 0, DIM, BB * NN, DIM, DIM, ep);
    }
}
